// SpikeEncoder_3435973836970
// MI455X (gfx1250) — matrix operand order repaired and checked by reading, not compiled, not run on hardware
//
#include <hip/hip_runtime.h>
#include <hip/hip_bf16.h>

// ---------------------------------------------------------------------------
// SNN forward, fused for MI455X (gfx1250, wave32, WMMA f16 16x16x32).
//   B=8192, D_in=1024, D_h=512, D_out=256, 25 steps, beta=0.9, thresh=1.0
// Design: cur1 GEMM once (f16 WMMA, f32 acc); persistent fused step kernel
// keeps mem1/cur1 in VGPRs, spk1 in LDS, mem2 in WMMA accumulators; only
// HBM traffic per step is the mandatory 16.8 MB of spk2/mem2 output (NT-hint).
// ---------------------------------------------------------------------------

#define B_SZ   8192
#define D_IN   1024
#define D_H    512
#define D_OUT  256
#define NSTEP  25
#define BETA   0.9f
#define LDP    520   // padded LDS row stride (halves) for spk1 tile

typedef __attribute__((ext_vector_type(16))) _Float16 v16h;
typedef __attribute__((ext_vector_type(8)))  _Float16 half8;
typedef __attribute__((ext_vector_type(8)))  float    v8f;

union V16 { v16h v; half8 h[2]; };

// Load 16 f16 from two 16-byte-aligned chunks into a WMMA fragment register set.
__device__ __forceinline__ v16h load2x16(const _Float16* p0, const _Float16* p1) {
    V16 u;
    u.h[0] = *reinterpret_cast<const half8*>(p0);
    u.h[1] = *reinterpret_cast<const half8*>(p1);
    return u.v;
}

// Load 8 f32 (16-byte aligned) and convert to 8 f16 (v_cvt_pk_f16_f32 pairs).
__device__ __forceinline__ half8 cvt8(const float* p) {
    float4 a = *reinterpret_cast<const float4*>(p);
    float4 b = *reinterpret_cast<const float4*>(p + 4);
    half8 h;
    h[0] = (_Float16)a.x; h[1] = (_Float16)a.y; h[2] = (_Float16)a.z; h[3] = (_Float16)a.w;
    h[4] = (_Float16)b.x; h[5] = (_Float16)b.y; h[6] = (_Float16)b.z; h[7] = (_Float16)b.w;
    return h;
}

// ---------------------------------------------------------------------------
// Kernel A: bulk f32 -> f16 conversion, 8 elements/thread, b128 in / b128 out.
// Grid must be sized exactly count / (256*8).
// ---------------------------------------------------------------------------
__global__ __launch_bounds__(256) void cvt_f16_kernel(const float* __restrict__ src,
                                                      _Float16* __restrict__ dst) {
    size_t i = ((size_t)blockIdx.x * 256 + threadIdx.x) * 8;
    *reinterpret_cast<half8*>(dst + i) = cvt8(src + i);
}

// ---------------------------------------------------------------------------
// Kernel B: cur1 = x @ W1^T + b1   (f16 WMMA, f32 accumulate, pre-cvt inputs)
//   grid (64, 8), 256 threads (8 waves). Block tile: 128 rows x 64 cols.
//   Wave w -> m-tile w; 4 n-tiles per wave; K loop 1024/32 = 32 iters.
// ---------------------------------------------------------------------------
__global__ __launch_bounds__(256) void gemm1_kernel(const _Float16* __restrict__ xh,
                                                    const _Float16* __restrict__ W1h,
                                                    const float* __restrict__ b1,
                                                    float* __restrict__ cur1) {
    const int tid  = threadIdx.x;
    const int wave = tid >> 5, lane = tid & 31;
    const int hlf  = lane >> 4, ml = lane & 15;
    const int blockM = blockIdx.x * 128;
    const int blockN = blockIdx.y * 64;

    const _Float16* xrow = xh + (size_t)(blockM + wave * 16 + ml) * D_IN;
    const _Float16* w0 = W1h + (size_t)(blockN +  0 + ml) * D_IN;
    const _Float16* w1 = W1h + (size_t)(blockN + 16 + ml) * D_IN;
    const _Float16* w2 = W1h + (size_t)(blockN + 32 + ml) * D_IN;
    const _Float16* w3 = W1h + (size_t)(blockN + 48 + ml) * D_IN;

    v8f acc0 = {}, acc1 = {}, acc2 = {}, acc3 = {};

    for (int k0 = 0; k0 < D_IN; k0 += 32) {
        // A fragment: K = 8*hlf + (i&7) + 16*(i>>3)
        v16h a = load2x16(xrow + k0 + 8 * hlf, xrow + k0 + 16 + 8 * hlf);
        const int kb = k0 + 8 * hlf;                    // B fragment: K = 8*hlf + (i&7) + 16*(i>>3)
        v16h b0 = load2x16(w0 + kb, w0 + kb + 16);
        v16h b1f = load2x16(w1 + kb, w1 + kb + 16);
        v16h b2f = load2x16(w2 + kb, w2 + kb + 16);
        v16h b3f = load2x16(w3 + kb, w3 + kb + 16);
        acc0 = __builtin_amdgcn_wmma_f32_16x16x32_f16(false, a, false, b0,  (short)0, acc0, false, false);
        acc1 = __builtin_amdgcn_wmma_f32_16x16x32_f16(false, a, false, b1f, (short)0, acc1, false, false);
        acc2 = __builtin_amdgcn_wmma_f32_16x16x32_f16(false, a, false, b2f, (short)0, acc2, false, false);
        acc3 = __builtin_amdgcn_wmma_f32_16x16x32_f16(false, a, false, b3f, (short)0, acc3, false, false);
    }

    const int r0 = blockM + wave * 16 + 8 * hlf;        // C rows: 8*hlf + r
#pragma unroll
    for (int j = 0; j < 4; ++j) {
        const int n = blockN + j * 16 + ml;
        const float bias = b1[n];
        const v8f acc = (j == 0) ? acc0 : (j == 1) ? acc1 : (j == 2) ? acc2 : acc3;
#pragma unroll
        for (int r = 0; r < 8; ++r)
            cur1[(size_t)(r0 + r) * D_H + n] = acc[r] + bias;
    }
}

// ---------------------------------------------------------------------------
// Kernel C: fused 25-step SNN. One block = 32 batch rows, 512 threads (16 waves).
//   mem1/cur1 tiles in registers, spk1 via LDS (padded), mem2 in WMMA acc regs.
//   Wave w: m-tile = w&1, n-tiles = (w>>1)*2 + {0,1}  (covers 32 x 256 outputs).
// ---------------------------------------------------------------------------
__global__ __launch_bounds__(512) void snn_steps_kernel(const float* __restrict__ cur1,
                                                        const _Float16* __restrict__ W2h,
                                                        const float* __restrict__ b2,
                                                        float* __restrict__ out) {
    __shared__ _Float16 spk1[32 * LDP];                 // ~33 KB LDS

    const int tid  = threadIdx.x;                       // 0..511
    const int wave = tid >> 5, lane = tid & 31;
    const int hlf  = lane >> 4, nl = lane & 15;
    const int row0 = blockIdx.x * 32;

    // Per-thread state: column = tid, rows row0..row0+31. One-shot reads -> NT.
    float c1r[32], m1[32];
#pragma unroll
    for (int e = 0; e < 32; ++e) {
        c1r[e] = __builtin_nontemporal_load(&cur1[(size_t)(row0 + e) * D_H + tid]);
        m1[e]  = 0.0f;
    }

    const int mtile = wave & 1;
    const int n0    = (wave >> 1) * 32 + nl;            // first n-tile column for this lane
    const float bias0 = b2[n0];
    const float bias1 = b2[n0 + 16];

    v8f mem2a = {}, mem2b = {};

    float* spk_out = out;
    float* mem_out = out + (size_t)NSTEP * B_SZ * D_OUT;

    const _Float16* arow = spk1 + (size_t)(mtile * 16 + nl) * LDP;
    const _Float16* w2r0 = W2h + (size_t)n0 * D_H;
    const _Float16* w2r1 = W2h + (size_t)(n0 + 16) * D_H;

    for (int t = 0; t < NSTEP; ++t) {
        __syncthreads();                                // prior step's readers done

        // Layer-1 leaky update + spike (binary f16) into LDS.
#pragma unroll
        for (int e = 0; e < 32; ++e) {
            float m = m1[e];
            m = BETA * m + c1r[e] - ((m > 1.0f) ? 1.0f : 0.0f);
            m1[e] = m;
            spk1[e * LDP + tid] = (m > 1.0f) ? (_Float16)1.0f : (_Float16)0.0f;
        }
        __syncthreads();

        // cur2 tile = spk1 @ W2^T : 16 K-steps of 16x16x32 f16 WMMA, 2 n-tiles.
        v8f acc0 = {}, acc1 = {};
#pragma unroll
        for (int kt = 0; kt < 16; ++kt) {
            const int k0 = kt * 32;
            v16h a = load2x16(arow + k0 + 8 * hlf, arow + k0 + 16 + 8 * hlf);
            const _Float16* p0 = w2r0 + k0 + 8 * hlf;
            const _Float16* p1 = w2r1 + k0 + 8 * hlf;
            v16h b0 = load2x16(p0, p0 + 16);
            v16h b1f = load2x16(p1, p1 + 16);
            acc0 = __builtin_amdgcn_wmma_f32_16x16x32_f16(false, a, false, b0,
                                                          (short)0, acc0, false, false);
            acc1 = __builtin_amdgcn_wmma_f32_16x16x32_f16(false, a, false, b1f,
                                                          (short)0, acc1, false, false);
        }

        // Layer-2 leaky update; record spk2/mem2 with NT stores (write-once data).
        const size_t base = (size_t)t * B_SZ * D_OUT;
        const int grow0 = row0 + mtile * 16 + 8 * hlf;
#pragma unroll
        for (int j = 0; j < 2; ++j) {
            v8f& m2  = (j == 0) ? mem2a : mem2b;
            v8f  a2  = (j == 0) ? acc0 : acc1;
            const float bias = (j == 0) ? bias0 : bias1;
            const int n = n0 + j * 16;
#pragma unroll
            for (int r = 0; r < 8; ++r) {
                float m = m2[r];
                m = BETA * m + a2[r] + bias - ((m > 1.0f) ? 1.0f : 0.0f);
                m2[r] = m;
                const size_t off = base + (size_t)(grow0 + r) * D_OUT + n;
                __builtin_nontemporal_store((m > 1.0f) ? 1.0f : 0.0f, &spk_out[off]);
                __builtin_nontemporal_store(m, &mem_out[off]);
            }
        }
    }
}

// ---------------------------------------------------------------------------
// Host launcher.
// Workspace layout: [cur1 f32 16.8MB][x f16 16.8MB][W1 f16 1MB][W2 f16 256KB]
// ---------------------------------------------------------------------------
extern "C" void kernel_launch(void* const* d_in, const int* in_sizes, int n_in,
                              void* d_out, int out_size, void* d_ws, size_t ws_size,
                              hipStream_t stream) {
    (void)in_sizes; (void)n_in; (void)out_size; (void)ws_size;
    const float* x  = (const float*)d_in[0];
    const float* W1 = (const float*)d_in[1];
    const float* b1 = (const float*)d_in[2];
    const float* W2 = (const float*)d_in[3];
    const float* b2 = (const float*)d_in[4];
    float* out = (float*)d_out;

    char* ws = (char*)d_ws;
    float*    cur1 = (float*)ws;                                   ws += (size_t)B_SZ * D_H * sizeof(float);
    _Float16* xh   = (_Float16*)ws;                                ws += (size_t)B_SZ * D_IN * sizeof(_Float16);
    _Float16* W1h  = (_Float16*)ws;                                ws += (size_t)D_H * D_IN * sizeof(_Float16);
    _Float16* W2h  = (_Float16*)ws;

    cvt_f16_kernel<<<dim3((B_SZ * D_IN) / 2048), 256, 0, stream>>>(x, xh);
    cvt_f16_kernel<<<dim3((D_H * D_IN) / 2048), 256, 0, stream>>>(W1, W1h);
    cvt_f16_kernel<<<dim3((D_OUT * D_H) / 2048), 256, 0, stream>>>(W2, W2h);
    gemm1_kernel<<<dim3(B_SZ / 128, D_H / 64), 256, 0, stream>>>(xh, W1h, b1, cur1);
    snn_steps_kernel<<<dim3(B_SZ / 32), 512, 0, stream>>>(cur1, W2h, b2, out);
}
